// SPINN_59115929862455
// MI455X (gfx1250) — matrix operand order repaired and checked by reading, not compiled, not run on hardware
//
#include <hip/hip_runtime.h>

// ---------------- types / helpers ----------------
typedef __attribute__((ext_vector_type(16))) __bf16 v16bf;
typedef __attribute__((ext_vector_type(8)))  float  v8f;

union FragBf16 { v16bf v; uint4 q[2]; };

__device__ __forceinline__ unsigned short f32_to_bf16(float f) {
    union { float f; unsigned int u; } v; v.f = f;
    unsigned int r = v.u + 0x7fffu + ((v.u >> 16) & 1u);   // round-to-nearest-even
    return (unsigned short)(r >> 16);
}
__device__ __forceinline__ float sigm(float x) { return 1.0f / (1.0f + __expf(-x)); }

// problem constants
#define S_      303
#define SP_     320      // padded K / stride for h,c buffers and S-wide outputs (10*32)
#define KFC_    608      // 606 -> 608 (19 * 32)
#define KG_     928      // 909 -> 928 (29 * 32)
#define GP_     1536     // 1515 -> 1536 (48 * 32)
#define B_      128
#define NNODES_ 1023
#define LEAF_   512

// ---------------- WMMA GEMM: Y = X @ W^T + bias ----------------
// X: bf16 (rows, LDX) row-major; W: bf16 (ncols, LDW) row-major (B = W^T).
// Register-blocked: one wave computes a 64x32 macro-tile (4 row-tiles x 2 col-tiles):
// per 32-K chunk: 4 A-frags + 2 B-frags (6 x b128-pair loads) feed 8 WMMAs.
// Strides & K are compile-time so row offsets fold into load immediates;
// only 2 live base pointers -> no spills. rows % 64 == 0; ncols % 32 == 0.
template <int LDX, int LDW, int K, int LDY>
__global__ __launch_bounds__(256, 1)
void spinn_gemm_bf16(const unsigned short* __restrict__ X,
                     const unsigned short* __restrict__ W,
                     const float* __restrict__ bias,
                     float* __restrict__ Y,
                     int rowBlocks, int colBlocks) {
    int gw = (int)((blockIdx.x * blockDim.x + threadIdx.x) >> 5);
    if (gw >= rowBlocks * colBlocks) return;        // uniform across wave
    int rb = gw / colBlocks, cb = gw - rb * colBlocks;
    int lane = (int)(threadIdx.x & 31);
    int m = lane & 15, half = lane >> 4;

    // per-lane base pointers; all tile offsets are compile-time immediates
    const unsigned short* Ap = X + (size_t)(rb * 64 + m) * LDX + half * 8;
    const unsigned short* Bp = W + (size_t)(cb * 32 + m) * LDW + half * 8;

    v8f acc[4][2];
#pragma unroll
    for (int i = 0; i < 4; ++i)
#pragma unroll
        for (int j = 0; j < 2; ++j) acc[i][j] = (v8f){};

    for (int k0 = 0; k0 < K; k0 += 32) {
        FragBf16 a[4], b[2];
#pragma unroll
        for (int i = 0; i < 4; ++i) {
            a[i].q[0] = *(const uint4*)(Ap + i * 16 * LDX);       // k = k0+half*8+t
            a[i].q[1] = *(const uint4*)(Ap + i * 16 * LDX + 16);  // k = k0+16+half*8+t
        }
#pragma unroll
        for (int j = 0; j < 2; ++j) {
            b[j].q[0] = *(const uint4*)(Bp + j * 16 * LDW);       // k = k0+half*8+t
            b[j].q[1] = *(const uint4*)(Bp + j * 16 * LDW + 16);
        }
        // speculative prefetch of next chunk (silently dropped if OOB)
        __builtin_prefetch(Ap + 32, 0, 0);
        __builtin_prefetch(Bp + 32, 0, 0);
#pragma unroll
        for (int i = 0; i < 4; ++i)
#pragma unroll
            for (int j = 0; j < 2; ++j)
                acc[i][j] = __builtin_amdgcn_wmma_f32_16x16x32_bf16(
                    false, a[i].v, false, b[j].v, (short)0, acc[i][j], false, false);
        Ap += 32;
        Bp += 32;
    }

#pragma unroll
    for (int j = 0; j < 2; ++j) {
        int col = cb * 32 + j * 16 + m;
        float bv = bias[col];
#pragma unroll
        for (int i = 0; i < 4; ++i) {
            float* ybase = Y + (size_t)(rb * 64 + i * 16 + half * 8) * LDY + col;
#pragma unroll
            for (int v = 0; v < 8; ++v) ybase[(size_t)v * LDY] = acc[i][j][v] + bv;
        }
    }
}

// ---------------- pack / convert kernels ----------------
__global__ void spinn_pack_weight(const float* __restrict__ src, int R, int C,
                                  unsigned short* __restrict__ dst, int Rp, int Cp) {
    long idx = (long)blockIdx.x * blockDim.x + threadIdx.x;
    if (idx >= (long)Rp * Cp) return;
    int r = (int)(idx / Cp), c = (int)(idx - (long)r * Cp);
    float v = (r < R && c < C) ? src[(size_t)r * C + c] : 0.0f;
    dst[idx] = f32_to_bf16(v);
}

__global__ void spinn_pack_gate_weight(const float* __restrict__ Wx,
                                       const float* __restrict__ Ul,
                                       const float* __restrict__ Ur,
                                       unsigned short* __restrict__ dst) {
    long idx = (long)blockIdx.x * blockDim.x + threadIdx.x;
    if (idx >= (long)GP_ * KG_) return;
    int n = (int)(idx / KG_), k = (int)(idx - (long)n * KG_);
    float v = 0.0f;
    if (n < 5 * S_) {
        if (k < S_)            v = Wx[(size_t)n * S_ + k];
        else if (k < 2 * S_)   v = Ul[(size_t)n * S_ + (k - S_)];
        else if (k < 3 * S_)   v = Ur[(size_t)n * S_ + (k - 2 * S_)];
    }
    dst[idx] = f32_to_bf16(v);
}

__global__ void spinn_pack_bias(const float* __restrict__ src, int n,
                                float* __restrict__ dst, int np) {
    int i = blockIdx.x * blockDim.x + threadIdx.x;
    if (i < np) dst[i] = (i < n) ? src[i] : 0.0f;
}

__global__ void spinn_pack_leafX(const float* __restrict__ nf,
                                 unsigned short* __restrict__ dst) {
    long idx = (long)blockIdx.x * blockDim.x + threadIdx.x;
    if (idx >= (long)B_ * LEAF_ * SP_) return;
    int r = (int)(idx / SP_), s = (int)(idx - (long)r * SP_);
    int b = r / LEAF_, j = r - b * LEAF_;
    float v = (s < S_) ? nf[((size_t)b * NNODES_ + (LEAF_ - 1) + j) * S_ + s] : 0.0f;
    dst[idx] = f32_to_bf16(v);
}

__global__ void spinn_tanh_to_bf16(const float* __restrict__ Y, int rows,
                                   unsigned short* __restrict__ h) {
    long idx = (long)blockIdx.x * blockDim.x + threadIdx.x;
    if (idx >= (long)rows * SP_) return;
    int r = (int)(idx / SP_), s = (int)(idx - (long)r * SP_);
    float v = (s < S_) ? tanhf(Y[(size_t)r * SP_ + s]) : 0.0f;
    h[idx] = f32_to_bf16(v);
}

__global__ void spinn_tanh_to_f32(const float* __restrict__ Y, int rows,
                                  float* __restrict__ c) {
    long idx = (long)blockIdx.x * blockDim.x + threadIdx.x;
    if (idx >= (long)rows * SP_) return;
    int r = (int)(idx / SP_), s = (int)(idx - (long)r * SP_);
    c[idx] = (s < S_) ? tanhf(Y[(size_t)r * SP_ + s]) : 0.0f;
}

// Xcat row (b,j) = [nf(left child) | nf(right child)] padded to KFC_
__global__ void spinn_pack_xcat(const float* __restrict__ nf, int cnt, int cstart,
                                unsigned short* __restrict__ dst, int rows) {
    long idx = (long)blockIdx.x * blockDim.x + threadIdx.x;
    if (idx >= (long)rows * KFC_) return;
    int r = (int)(idx / KFC_), t = (int)(idx - (long)r * KFC_);
    int b = r / cnt, j = r - b * cnt;
    float v = 0.0f;
    if (t < S_)            v = nf[((size_t)b * NNODES_ + cstart + 2 * j) * S_ + t];
    else if (t < 2 * S_)   v = nf[((size_t)b * NNODES_ + cstart + 2 * j + 1) * S_ + (t - S_)];
    dst[idx] = f32_to_bf16(v);
}

// Xg row (b,j) = [bf16(x) | h_l | h_r] padded to KG_
__global__ void spinn_pack_xg(const float* __restrict__ x, const unsigned short* __restrict__ hprev,
                              int cnt, unsigned short* __restrict__ dst, int rows) {
    long idx = (long)blockIdx.x * blockDim.x + threadIdx.x;
    if (idx >= (long)rows * KG_) return;
    int r = (int)(idx / KG_), t = (int)(idx - (long)r * KG_);
    int b = r / cnt, j = r - b * cnt;
    size_t hl = ((size_t)b * (2 * cnt) + 2 * j) * SP_;
    unsigned short v = 0;
    if (t < S_)            v = f32_to_bf16(x[(size_t)r * SP_ + t]);
    else if (t < 2 * S_)   v = hprev[hl + (t - S_)];
    else if (t < 3 * S_)   v = hprev[hl + SP_ + (t - 2 * S_)];
    dst[idx] = v;
}

__global__ void spinn_lstm_cell(const float* __restrict__ gates, const float* __restrict__ cprev,
                                int cnt, float* __restrict__ cnew,
                                unsigned short* __restrict__ hnew, int rows) {
    long idx = (long)blockIdx.x * blockDim.x + threadIdx.x;
    if (idx >= (long)rows * SP_) return;
    int r = (int)(idx / SP_), s = (int)(idx - (long)r * SP_);
    if (s >= S_) { cnew[idx] = 0.0f; hnew[idx] = 0; return; }
    int b = r / cnt, j = r - b * cnt;
    size_t gb = (size_t)r * GP_;
    float gi = gates[gb + s];
    float fl = gates[gb + S_ + s];
    float fr = gates[gb + 2 * S_ + s];
    float go = gates[gb + 3 * S_ + s];
    float gg = gates[gb + 4 * S_ + s];
    size_t cl = ((size_t)b * (2 * cnt) + 2 * j) * SP_ + s;
    float c = sigm(gi) * tanhf(gg) + sigm(fl) * cprev[cl] + sigm(fr) * cprev[cl + SP_];
    cnew[idx] = c;
    hnew[idx] = f32_to_bf16(sigm(go) * tanhf(c));
}

// out[b] = sigmoid(b2 + sum_s relu(z1[b,s]) * Wm2[s])
__global__ void spinn_mlp_final(const float* __restrict__ z1, const float* __restrict__ Wm2,
                                const float* __restrict__ b2, float* __restrict__ out) {
    int b = threadIdx.x;
    if (b >= B_) return;
    float acc = 0.0f;
    for (int s = 0; s < S_; ++s) {
        float z = z1[(size_t)b * SP_ + s];
        acc += (z > 0.0f ? z : 0.0f) * Wm2[s];
    }
    out[b] = sigm(acc + b2[0]);
}

// ---------------- host launch ----------------
static inline dim3 g1(long n) { return dim3((unsigned)((n + 255) / 256)); }

extern "C" void kernel_launch(void* const* d_in, const int* in_sizes, int n_in,
                              void* d_out, int out_size, void* d_ws, size_t ws_size,
                              hipStream_t stream) {
    (void)in_sizes; (void)n_in; (void)out_size; (void)ws_size;
    const float* nf   = (const float*)d_in[0];
    const float* Wlh  = (const float*)d_in[1];
    const float* blh  = (const float*)d_in[2];
    const float* Wlc  = (const float*)d_in[3];
    const float* blc  = (const float*)d_in[4];
    const float* Wfc  = (const float*)d_in[5];
    const float* bfc  = (const float*)d_in[6];
    const float* Wx   = (const float*)d_in[7];
    const float* Ul   = (const float*)d_in[8];
    const float* Ur   = (const float*)d_in[9];
    const float* blst = (const float*)d_in[10];
    const float* Wm1  = (const float*)d_in[11];
    const float* bm1  = (const float*)d_in[12];
    const float* Wm2  = (const float*)d_in[13];
    const float* bm2  = (const float*)d_in[14];
    float* out = (float*)d_out;

    char* ws = (char*)d_ws;
    size_t off = 0;
    auto alloc = [&](size_t bytes) -> char* {
        char* p = ws + off;
        off = (off + bytes + 255) & ~(size_t)255;
        return p;
    };
    unsigned short* pWh  = (unsigned short*)alloc((size_t)SP_ * SP_ * 2);
    unsigned short* pWc  = (unsigned short*)alloc((size_t)SP_ * SP_ * 2);
    unsigned short* pWfc = (unsigned short*)alloc((size_t)SP_ * KFC_ * 2);
    unsigned short* pWg  = (unsigned short*)alloc((size_t)GP_ * KG_ * 2);
    unsigned short* pWm1 = (unsigned short*)alloc((size_t)SP_ * SP_ * 2);
    float* pbh   = (float*)alloc(SP_ * 4);
    float* pbc   = (float*)alloc(SP_ * 4);
    float* pbfc  = (float*)alloc(SP_ * 4);
    float* pblst = (float*)alloc(GP_ * 4);
    float* pbm1  = (float*)alloc(SP_ * 4);
    unsigned short* leafX = (unsigned short*)alloc((size_t)B_ * LEAF_ * SP_ * 2);
    float* Ybuf  = (float*)alloc((size_t)B_ * LEAF_ * SP_ * 4);
    unsigned short* hA = (unsigned short*)alloc((size_t)B_ * LEAF_ * SP_ * 2);
    unsigned short* hB = (unsigned short*)alloc((size_t)B_ * LEAF_ * SP_ * 2);
    float* cA = (float*)alloc((size_t)B_ * LEAF_ * SP_ * 4);
    float* cB = (float*)alloc((size_t)B_ * LEAF_ * SP_ * 4);
    unsigned short* Xcat = (unsigned short*)alloc((size_t)B_ * 256 * KFC_ * 2);
    unsigned short* Xg   = (unsigned short*)alloc((size_t)B_ * 256 * KG_ * 2);
    float* Gates = (float*)alloc((size_t)B_ * 256 * GP_ * 4);

    const int BT = 256;
    // --- pack weights / biases (bf16, zero-padded) ---
    spinn_pack_weight<<<g1((long)SP_ * SP_),  BT, 0, stream>>>(Wlh, S_, S_,      pWh,  SP_, SP_);
    spinn_pack_weight<<<g1((long)SP_ * SP_),  BT, 0, stream>>>(Wlc, S_, S_,      pWc,  SP_, SP_);
    spinn_pack_weight<<<g1((long)SP_ * KFC_), BT, 0, stream>>>(Wfc, S_, 2 * S_,  pWfc, SP_, KFC_);
    spinn_pack_weight<<<g1((long)SP_ * SP_),  BT, 0, stream>>>(Wm1, S_, S_,      pWm1, SP_, SP_);
    spinn_pack_gate_weight<<<g1((long)GP_ * KG_), BT, 0, stream>>>(Wx, Ul, Ur, pWg);
    spinn_pack_bias<<<g1(SP_), BT, 0, stream>>>(blh,  S_,      pbh,   SP_);
    spinn_pack_bias<<<g1(SP_), BT, 0, stream>>>(blc,  S_,      pbc,   SP_);
    spinn_pack_bias<<<g1(SP_), BT, 0, stream>>>(bfc,  S_,      pbfc,  SP_);
    spinn_pack_bias<<<g1(GP_), BT, 0, stream>>>(blst, 5 * S_,  pblst, GP_);
    spinn_pack_bias<<<g1(SP_), BT, 0, stream>>>(bm1,  S_,      pbm1,  SP_);

    // --- leaves: h = tanh(X Wh^T + bh), c = tanh(X Wc^T + bc) ---
    const int leafRows = B_ * LEAF_;
    spinn_pack_leafX<<<g1((long)leafRows * SP_), BT, 0, stream>>>(nf, leafX);
    {
        long waves = (long)(leafRows / 64) * (SP_ / 32);
        spinn_gemm_bf16<SP_, SP_, SP_, SP_><<<g1(waves * 32), BT, 0, stream>>>(
            leafX, pWh, pbh, Ybuf, leafRows / 64, SP_ / 32);
        spinn_tanh_to_bf16<<<g1((long)leafRows * SP_), BT, 0, stream>>>(Ybuf, leafRows, hA);
        spinn_gemm_bf16<SP_, SP_, SP_, SP_><<<g1(waves * 32), BT, 0, stream>>>(
            leafX, pWc, pbc, Ybuf, leafRows / 64, SP_ / 32);
        spinn_tanh_to_f32<<<g1((long)leafRows * SP_), BT, 0, stream>>>(Ybuf, leafRows, cA);
    }

    // --- tree levels, bottom-up ---
    unsigned short* h_prev = hA; float* c_prev = cA;
    unsigned short* h_cur  = hB; float* c_cur  = cB;
    for (int L = 8; L >= 0; --L) {
        int cnt = 1 << L;
        int rows = B_ * cnt;
        int cstart = (1 << (L + 1)) - 1;
        spinn_pack_xcat<<<g1((long)rows * KFC_), BT, 0, stream>>>(nf, cnt, cstart, Xcat, rows);
        long wavesX = (long)(rows / 64) * (SP_ / 32);
        spinn_gemm_bf16<KFC_, KFC_, KFC_, SP_><<<g1(wavesX * 32), BT, 0, stream>>>(
            Xcat, pWfc, pbfc, Ybuf, rows / 64, SP_ / 32);
        spinn_pack_xg<<<g1((long)rows * KG_), BT, 0, stream>>>(Ybuf, h_prev, cnt, Xg, rows);
        long wavesG = (long)(rows / 64) * (GP_ / 32);
        spinn_gemm_bf16<KG_, KG_, KG_, GP_><<<g1(wavesG * 32), BT, 0, stream>>>(
            Xg, pWg, pblst, Gates, rows / 64, GP_ / 32);
        spinn_lstm_cell<<<g1((long)rows * SP_), BT, 0, stream>>>(Gates, c_prev, cnt, c_cur, h_cur, rows);
        unsigned short* th = h_prev; h_prev = h_cur; h_cur = th;
        float* tc = c_prev; c_prev = c_cur; c_cur = tc;
    }

    // --- MLP head: z1 = h_root Wm1^T + bm1 (relu inside final), out = sigmoid(relu(z1) Wm2^T + b2)
    {
        long waves = (long)(B_ / 64) * (SP_ / 32);
        spinn_gemm_bf16<SP_, SP_, SP_, SP_><<<g1(waves * 32), BT, 0, stream>>>(
            h_prev, pWm1, pbm1, Ybuf, B_ / 64, SP_ / 32);
        spinn_mlp_final<<<1, 128, 0, stream>>>(Ybuf, Wm2, bm2, out);
    }
}
